// NanoGPTCausalSelfAttention_10746008175367
// MI455X (gfx1250) — matrix operand order repaired and checked by reading, not compiled, not run on hardware
//
#include <hip/hip_runtime.h>
#include <hip/hip_bf16.h>
#include <cstdint>

// ---------------------------------------------------------------------------
// NanoGPT causal self-attention forward, CDNA5 (gfx1250), wave32 + WMMA f16,
// async global->LDS staging (ASYNCcnt) for GEMM weight tiles.
// B=2, T=2048, C=1024, H=16, D=64.
// ---------------------------------------------------------------------------

typedef __attribute__((ext_vector_type(16))) _Float16 v16h;
typedef __attribute__((ext_vector_type(8)))  _Float16 v8h;
typedef __attribute__((ext_vector_type(8)))  float    v8f;

union V16H { v16h v; v8h h[2]; };

#define CONST_B 2
#define CONST_T 2048
#define CONST_C 1024
#define CONST_H 16
#define CONST_D 64
#define CONST_M (CONST_B * CONST_T)   // 4096 token rows

static __device__ inline v8f wmma_f16(v16h a, v16h b, v8f c) {
  // D = A(16x32 f16) * B(32x16 f16) + C(16x16 f32)
  return __builtin_amdgcn_wmma_f32_16x16x32_f16(
      /*neg_a=*/false, a, /*neg_b=*/false, b,
      /*c_mod=*/(short)0, c, /*reuse_a=*/false, /*reuse_b=*/false);
}

// A-operand (16x32 f16) fragment from row-major memory, leading dim lda.
// Lane L holds row M=L%16; K = 16 contiguous values starting at (L<16 ? 0 : 16).
static __device__ inline v16h load_a_frag(const _Float16* __restrict__ base, int lda) {
  int lane = threadIdx.x & 31;
  int row  = lane & 15;
  int koff = (lane & 16) ? 16 : 0;
  const _Float16* p = base + (size_t)row * lda + koff;
  V16H u;
  u.h[0] = *(const v8h*)(p);
  u.h[1] = *(const v8h*)(p + 8);
  return u.v;
}

// B-operand (32x16 f16) where B^T is row-major with leading dim ldw (source
// row n holds column n of B contiguously in K). Lane L: col N=L%16,
// K = 16 contiguous values starting at (L<16 ? 0 : 16).
static __device__ inline v16h load_b_frag(const _Float16* __restrict__ base, int ldw) {
  int lane = threadIdx.x & 31;
  int col  = lane & 15;
  int koff = (lane & 16) ? 16 : 0;
  return *(const v16h*)(base + (size_t)col * ldw + koff);
}

// ---------------------------------------------------------------------------
// f32 -> f16 cast (memory bound, negligible)
// ---------------------------------------------------------------------------
__global__ void cast_f32_to_f16(const float* __restrict__ in,
                                _Float16* __restrict__ out, int n) {
  int i = blockIdx.x * blockDim.x + threadIdx.x;
  if (i < n) out[i] = (_Float16)in[i];
}

// ---------------------------------------------------------------------------
// LDS staging geometry for GEMM weight tiles: 128 rows x 32 halfs per k-step,
// row stride 56 halfs (112 B: 16B-aligned async b128 chunks; row*28 mod 64
// covers 16 distinct LDS bank groups for the fragment reads).
// ---------------------------------------------------------------------------
#define BSTRIDE 56

static __device__ inline void async_b128_to_lds(unsigned lds_addr, unsigned goff,
                                                const _Float16* sbase) {
  asm volatile("global_load_async_to_lds_b128 %0, %1, %2"
               :: "v"(lds_addr), "v"(goff), "s"(sbase)
               : "memory");
}

static __device__ inline void wait_async0() {
  asm volatile("s_wait_asynccnt 0x0" ::: "memory");
}

// ---------------------------------------------------------------------------
// QKV GEMM: qkv[M=4096, N=3072] = Xh[M,1024] @ W_attn^T + b_attn
// Block: 256 threads = 8 waves, tile 128x128, wave tile 32x64, K step 32.
// Weight tile double-buffered in LDS via async loads; A direct from global.
// Epilogue scatters to Q[B,H,T,D], K[B,H,T,D], Vt[B,H,D,T] as f16.
// ---------------------------------------------------------------------------
__global__ __launch_bounds__(256)
void qkv_gemm_kernel(const _Float16* __restrict__ Xh,
                     const _Float16* __restrict__ Wh,   // [3072,1024] f16
                     const float* __restrict__ bias,    // [3072]
                     _Float16* __restrict__ Qh,
                     _Float16* __restrict__ Kh,
                     _Float16* __restrict__ Vt) {
  __shared__ __align__(16) _Float16 Bs[2][128 * BSTRIDE];

  const int K = CONST_C;
  int tid  = threadIdx.x;
  int wave = tid >> 5;
  int lane = tid & 31;
  int wm = wave & 3;        // 4 waves over M
  int wn = wave >> 2;       // 2 waves over N
  int m0 = blockIdx.x * 128 + wm * 32;
  int nb = blockIdx.y * 128;          // block's N base
  int n0w = wn * 64;                  // wave's N offset within tile

  // Async-copy mapping: tile = 512 x 16B chunks, 2 chunks per thread.
  int c0 = tid * 2, c1 = c0 + 1;
  int row0 = c0 >> 2, sub0 = c0 & 3;
  int row1 = c1 >> 2, sub1 = c1 & 3;
  unsigned goff0 = (unsigned)(row0 * K + sub0 * 8) * 2u;       // bytes
  unsigned goff1 = (unsigned)(row1 * K + sub1 * 8) * 2u;
  unsigned lds0 = (unsigned)(uintptr_t)&Bs[0][0];
  unsigned lds1 = (unsigned)(uintptr_t)&Bs[1][0];
  unsigned loff0 = (unsigned)(row0 * BSTRIDE + sub0 * 8) * 2u;  // bytes
  unsigned loff1 = (unsigned)(row1 * BSTRIDE + sub1 * 8) * 2u;
  const _Float16* Wb = Wh + (size_t)nb * K;

  // prologue: stage k-step 0 into buffer 0
  async_b128_to_lds(lds0 + loff0, goff0, Wb);
  async_b128_to_lds(lds0 + loff1, goff1, Wb);

  v8f acc[2][4] = {};
  int cl = lane & 15;
  for (int k0 = 0; k0 < K; k0 += 32) {
    int cur = (k0 >> 5) & 1;
    wait_async0();        // own chunks of current tile landed
    __syncthreads();      // all waves' chunks visible; prev readers done
    if (k0 + 32 < K) {
      unsigned lb = cur ? lds0 : lds1;
      const _Float16* gb = Wb + (k0 + 32);
      async_b128_to_lds(lb + loff0, goff0, gb);
      async_b128_to_lds(lb + loff1, goff1, gb);
    }
    // hoist all fragment loads so ds/global latency is covered once
    v16h a0 = load_a_frag(Xh + (size_t)m0 * K + k0, K);
    v16h a1 = load_a_frag(Xh + (size_t)(m0 + 16) * K + k0, K);
    v16h bfrag[4];
#pragma unroll
    for (int j = 0; j < 4; ++j)
      bfrag[j] = load_b_frag(&Bs[cur][(size_t)(n0w + j * 16) * BSTRIDE], BSTRIDE);
#pragma unroll
    for (int j = 0; j < 4; ++j) {
      acc[0][j] = wmma_f16(a0, bfrag[j], acc[0][j]);
      acc[1][j] = wmma_f16(a1, bfrag[j], acc[1][j]);
    }
  }

  int rh = (lane >> 4) << 3;   // row offset 0 or 8
#pragma unroll
  for (int i = 0; i < 2; ++i) {
#pragma unroll
    for (int j = 0; j < 4; ++j) {
      int n = nb + n0w + j * 16 + cl;
      float bv = bias[n];
      int which = n >> 10;           // 0=Q 1=K 2=V
      int h = (n >> 6) & 15;
      int d = n & 63;
#pragma unroll
      for (int r = 0; r < 8; ++r) {
        int m = m0 + i * 16 + r + rh;
        int b_idx = m >> 11;
        int t = m & (CONST_T - 1);
        _Float16 hv = (_Float16)(acc[i][j][r] + bv);
        size_t bh = (size_t)(b_idx * CONST_H + h);
        if (which == 0)      Qh[(bh * CONST_T + t) * CONST_D + d] = hv;
        else if (which == 1) Kh[(bh * CONST_T + t) * CONST_D + d] = hv;
        else                 Vt[(bh * CONST_D + d) * CONST_T + t] = hv;
      }
    }
  }
}

// ---------------------------------------------------------------------------
// Flash attention: one wave per 16-query tile, 32 keys per iteration.
// S = Q @ K^T (4 WMMA), online softmax, P staged via LDS, O += P @ V (4 WMMA).
// V fragments are fetched at the top of the iteration so the global loads
// overlap the whole softmax VALU section.
// ---------------------------------------------------------------------------
#define P_STRIDE 40   // 16x32 f16 tile padded: bank-conflict-free A reload

__global__ __launch_bounds__(256)
void attn_kernel(const _Float16* __restrict__ Qh,
                 const _Float16* __restrict__ Kh,
                 const _Float16* __restrict__ Vt,
                 _Float16* __restrict__ Yh) {
  __shared__ __align__(16) _Float16 lds[8][16 * P_STRIDE];

  int wave = threadIdx.x >> 5;
  int lane = threadIdx.x & 31;
  int task = blockIdx.x * 8 + wave;        // 0..4095
  int qt = task & 127;                     // query tile within (b,h)
  int bh = task >> 7;                      // 0..31
  int q0 = qt << 4;

  const _Float16* Qb = Qh + (size_t)bh * CONST_T * CONST_D;
  const _Float16* Kb = Kh + (size_t)bh * CONST_T * CONST_D;
  const _Float16* Vb = Vt + (size_t)bh * CONST_D * CONST_T;

  v16h aq0 = load_a_frag(Qb + (size_t)q0 * CONST_D, CONST_D);
  v16h aq1 = load_a_frag(Qb + (size_t)q0 * CONST_D + 32, CONST_D);

  float mrow[8], lrow[8];
#pragma unroll
  for (int r = 0; r < 8; ++r) { mrow[r] = -1e30f; lrow[r] = 0.0f; }
  v8f o[4] = {};

  int cl = lane & 15;
  int rh = (lane >> 4) << 3;
  const float scale = 0.125f;              // 1/sqrt(64)
  int kend = q0 + 16;
  _Float16* P = lds[wave];

  for (int k0 = 0; k0 < kend; k0 += 32) {
    // ---- issue V fragment loads early (consumed after softmax)
    v16h bv[4];
#pragma unroll
    for (int dt = 0; dt < 4; ++dt) {
      int koff = (lane & 16) ? 16 : 0;
      bv[dt] = *(const v16h*)(Vb + (size_t)(dt * 16 + cl) * CONST_T + k0 + koff);
    }
    // ---- K fragments, then the 4 score WMMAs
    v16h bk[2][2];
#pragma unroll
    for (int t = 0; t < 2; ++t) {
      bk[t][0] = load_b_frag(Kb + (size_t)(k0 + t * 16) * CONST_D + 0, CONST_D);
      bk[t][1] = load_b_frag(Kb + (size_t)(k0 + t * 16) * CONST_D + 32, CONST_D);
    }
    v8f s[2] = {};
#pragma unroll
    for (int t = 0; t < 2; ++t) {
      s[t] = wmma_f16(aq0, bk[t][0], s[t]);
      s[t] = wmma_f16(aq1, bk[t][1], s[t]);
    }
    // ---- scale + causal mask + row max over 32 keys
    float chmax[8];
#pragma unroll
    for (int r = 0; r < 8; ++r) {
      int q = q0 + r + rh;
#pragma unroll
      for (int t = 0; t < 2; ++t) {
        int key = k0 + t * 16 + cl;
        float sv = s[t][r] * scale;
        s[t][r] = (key <= q) ? sv : -1e30f;
      }
      float cm = fmaxf(s[0][r], s[1][r]);
#pragma unroll
      for (int off = 1; off < 16; off <<= 1)
        cm = fmaxf(cm, __shfl_xor(cm, off, 32));
      chmax[r] = cm;
    }
    // ---- online softmax update, stage P into LDS
#pragma unroll
    for (int r = 0; r < 8; ++r) {
      float mnew = fmaxf(mrow[r], chmax[r]);
      float corr = __expf(mrow[r] - mnew);
      float p0 = __expf(s[0][r] - mnew);
      float p1 = __expf(s[1][r] - mnew);
      float rs = p0 + p1;
#pragma unroll
      for (int off = 1; off < 16; off <<= 1)
        rs += __shfl_xor(rs, off, 32);
      lrow[r] = lrow[r] * corr + rs;
      mrow[r] = mnew;
#pragma unroll
      for (int dt = 0; dt < 4; ++dt) o[dt][r] = o[dt][r] * corr;
      int row = r + rh;
      P[row * P_STRIDE + cl]      = (_Float16)p0;
      P[row * P_STRIDE + 16 + cl] = (_Float16)p1;
    }
    __builtin_amdgcn_wave_barrier();   // LDS in-order within wave
    v16h ap;
    {
      int koff = (lane & 16) ? 16 : 0;
      const _Float16* pp = P + cl * P_STRIDE + koff;
      V16H u;
      u.h[0] = *(const v8h*)(pp);
      u.h[1] = *(const v8h*)(pp + 8);
      ap = u.v;
    }
    __builtin_amdgcn_wave_barrier();
#pragma unroll
    for (int dt = 0; dt < 4; ++dt)
      o[dt] = wmma_f16(ap, bv[dt], o[dt]);
  }

  int b_idx = bh >> 4;
  int h = bh & 15;
#pragma unroll
  for (int r = 0; r < 8; ++r) {
    float inv = 1.0f / lrow[r];
    int q = q0 + r + rh;
    size_t rowoff = ((size_t)b_idx * CONST_T + q) * CONST_C + h * CONST_D;
#pragma unroll
    for (int dt = 0; dt < 4; ++dt)
      Yh[rowoff + dt * 16 + cl] = (_Float16)(o[dt][r] * inv);
  }
}

// ---------------------------------------------------------------------------
// Output projection: out[M=4096, N=1024] (f32) = Yh @ W_proj^T + b_proj
// Same async-staged structure as the QKV GEMM.
// ---------------------------------------------------------------------------
__global__ __launch_bounds__(256)
void proj_gemm_kernel(const _Float16* __restrict__ Yh,
                      const _Float16* __restrict__ Wh,   // [1024,1024] f16
                      const float* __restrict__ bias,    // [1024]
                      float* __restrict__ out) {
  __shared__ __align__(16) _Float16 Bs[2][128 * BSTRIDE];

  const int K = CONST_C;
  int tid  = threadIdx.x;
  int wave = tid >> 5;
  int lane = tid & 31;
  int wm = wave & 3;
  int wn = wave >> 2;
  int m0 = blockIdx.x * 128 + wm * 32;
  int nb = blockIdx.y * 128;
  int n0w = wn * 64;

  int c0 = tid * 2, c1 = c0 + 1;
  int row0 = c0 >> 2, sub0 = c0 & 3;
  int row1 = c1 >> 2, sub1 = c1 & 3;
  unsigned goff0 = (unsigned)(row0 * K + sub0 * 8) * 2u;
  unsigned goff1 = (unsigned)(row1 * K + sub1 * 8) * 2u;
  unsigned lds0 = (unsigned)(uintptr_t)&Bs[0][0];
  unsigned lds1 = (unsigned)(uintptr_t)&Bs[1][0];
  unsigned loff0 = (unsigned)(row0 * BSTRIDE + sub0 * 8) * 2u;
  unsigned loff1 = (unsigned)(row1 * BSTRIDE + sub1 * 8) * 2u;
  const _Float16* Wb = Wh + (size_t)nb * K;

  async_b128_to_lds(lds0 + loff0, goff0, Wb);
  async_b128_to_lds(lds0 + loff1, goff1, Wb);

  v8f acc[2][4] = {};
  int cl = lane & 15;
  for (int k0 = 0; k0 < K; k0 += 32) {
    int cur = (k0 >> 5) & 1;
    wait_async0();
    __syncthreads();
    if (k0 + 32 < K) {
      unsigned lb = cur ? lds0 : lds1;
      const _Float16* gb = Wb + (k0 + 32);
      async_b128_to_lds(lb + loff0, goff0, gb);
      async_b128_to_lds(lb + loff1, goff1, gb);
    }
    v16h a0 = load_a_frag(Yh + (size_t)m0 * K + k0, K);
    v16h a1 = load_a_frag(Yh + (size_t)(m0 + 16) * K + k0, K);
    v16h bfrag[4];
#pragma unroll
    for (int j = 0; j < 4; ++j)
      bfrag[j] = load_b_frag(&Bs[cur][(size_t)(n0w + j * 16) * BSTRIDE], BSTRIDE);
#pragma unroll
    for (int j = 0; j < 4; ++j) {
      acc[0][j] = wmma_f16(a0, bfrag[j], acc[0][j]);
      acc[1][j] = wmma_f16(a1, bfrag[j], acc[1][j]);
    }
  }

  int rh = (lane >> 4) << 3;
#pragma unroll
  for (int i = 0; i < 2; ++i) {
#pragma unroll
    for (int j = 0; j < 4; ++j) {
      int n = nb + n0w + j * 16 + cl;
      float bv = bias[n];
#pragma unroll
      for (int r = 0; r < 8; ++r) {
        int m = m0 + i * 16 + r + rh;
        out[(size_t)m * CONST_C + n] = acc[i][j][r] + bv;
      }
    }
  }
}

// ---------------------------------------------------------------------------
// Host-side launcher
// ---------------------------------------------------------------------------
extern "C" void kernel_launch(void* const* d_in, const int* in_sizes, int n_in,
                              void* d_out, int out_size, void* d_ws, size_t ws_size,
                              hipStream_t stream) {
  const float* x      = (const float*)d_in[0];  // [2,2048,1024]
  const float* W_attn = (const float*)d_in[1];  // [3072,1024]
  const float* b_attn = (const float*)d_in[2];  // [3072]
  const float* W_proj = (const float*)d_in[3];  // [1024,1024]
  const float* b_proj = (const float*)d_in[4];  // [1024]
  float* out = (float*)d_out;                   // [2,2048,1024]

  _Float16* ws = (_Float16*)d_ws;
  size_t off = 0;
  _Float16* Xh  = ws + off; off += (size_t)CONST_M * CONST_C;
  _Float16* Wah = ws + off; off += (size_t)3 * CONST_C * CONST_C;
  _Float16* Wph = ws + off; off += (size_t)CONST_C * CONST_C;
  const size_t bhtd = (size_t)CONST_B * CONST_H * CONST_T * CONST_D;
  _Float16* Qh = ws + off; off += bhtd;
  _Float16* Kh = ws + off; off += bhtd;
  _Float16* Vt = ws + off; off += bhtd;      // [B,H,D,T]
  _Float16* Yh = ws + off; off += (size_t)CONST_M * CONST_C;
  (void)ws_size; (void)n_in; (void)in_sizes; (void)out_size;

  {
    int n1 = CONST_M * CONST_C;
    cast_f32_to_f16<<<(n1 + 255) / 256, 256, 0, stream>>>(x, Xh, n1);
    int n2 = 3 * CONST_C * CONST_C;
    cast_f32_to_f16<<<(n2 + 255) / 256, 256, 0, stream>>>(W_attn, Wah, n2);
    int n3 = CONST_C * CONST_C;
    cast_f32_to_f16<<<(n3 + 255) / 256, 256, 0, stream>>>(W_proj, Wph, n3);
  }

  {
    dim3 grid(CONST_M / 128, (3 * CONST_C) / 128);  // 32 x 24
    qkv_gemm_kernel<<<grid, 256, 0, stream>>>(Xh, Wah, b_attn, Qh, Kh, Vt);
  }
  {
    int tasks = CONST_B * CONST_H * (CONST_T / 16);  // 4096 wave tasks
    attn_kernel<<<tasks / 8, 256, 0, stream>>>(Qh, Kh, Vt, Yh);
  }
  {
    dim3 grid(CONST_M / 128, CONST_C / 128);  // 32 x 8
    proj_gemm_kernel<<<grid, 256, 0, stream>>>(Yh, Wph, b_proj, out);
  }
}
